// TransformerModel_12120397710060
// MI455X (gfx1250) — matrix operand order repaired and checked by reading, not compiled, not run on hardware
//
#include <hip/hip_runtime.h>

// ---------------------------------------------------------------------------
// Model constants (from reference)
// ---------------------------------------------------------------------------
#define NN     8192          // total nodes
#define NDRUG  2048
#define NMETA  6144
#define ETR    131072
#define EVA    32768
#define ETOT   (ETR + EVA)   // 163840 edges
#define KSPLIT 4             // K-split factor for hop GEMM
#define KSEG   (NN / KSPLIT) // 2048 (multiple of 64 -> exact 2x-unrolled pipe)

// ---------------------------------------------------------------------------
// WMMA types / helpers (CDNA5 gfx1250, wave32)
// ---------------------------------------------------------------------------
typedef __attribute__((ext_vector_type(16))) __bf16 v16bf;
typedef __attribute__((ext_vector_type(8)))  __bf16 v8bf;
typedef __attribute__((ext_vector_type(8)))  float  v8f;

union V16 { v16bf v; v8bf h[2]; };

__device__ __forceinline__ v8f wmma_bf16(v16bf a, v16bf b, v8f c) {
  // D = A(16x32 bf16) * B(32x16 bf16) + C(16x16 f32)
  return __builtin_amdgcn_wmma_f32_16x16x32_bf16(false, a, false, b, (short)0, c,
                                                 false, false);
}

// Load a 16x32 bf16 A-tile chunk from a row-major [16 x stride] bf16 buffer.
// Layout: lanes 0-15 hold row=lane, K {0..15}; lanes 16-31 hold
// row=lane-16, K {16..31}.
__device__ __forceinline__ v16bf loadA_rowmajor(const __bf16* base, int stride,
                                                int kbase, int lmod, int lhalf) {
  const __bf16* p = base + lmod * stride + kbase + lhalf * 16;
  V16 u;
  u.h[0] = *(const v8bf*)p;
  u.h[1] = *(const v8bf*)(p + 8);
  return u.v;
}

__device__ __forceinline__ float wave_sum(float v) {
#pragma unroll
  for (int o = 16; o > 0; o >>= 1) v += __shfl_xor(v, o, 32);
  return v;
}
__device__ __forceinline__ float head_sum(float v) {  // reduce over 8-lane head
  v += __shfl_xor(v, 1, 32);
  v += __shfl_xor(v, 2, 32);
  v += __shfl_xor(v, 4, 32);
  return v;
}
__device__ __forceinline__ float gelu_tanh(float x) {
  float x3 = x * x * x;
  return 0.5f * x * (1.0f + tanhf(0.7978845608f * (x + 0.044715f * x3)));
}

// ---------------------------------------------------------------------------
// 1) f32 -> bf16 bulk convert (adjacency matrices; 64M elements each)
// ---------------------------------------------------------------------------
__global__ void k_f32_to_bf16(const float* __restrict__ in,
                              __bf16* __restrict__ out, size_t n) {
  size_t i = ((size_t)blockIdx.x * blockDim.x + threadIdx.x) * 4;
  if (i >= n) return;
  float4 v = *(const float4*)(in + i);
  union { __bf16 e[4]; uint2 u; } p;
  p.e[0] = (__bf16)v.x; p.e[1] = (__bf16)v.y;
  p.e[2] = (__bf16)v.z; p.e[3] = (__bf16)v.w;
  *(uint2*)(out + i) = p.u;
}

// ---------------------------------------------------------------------------
// 2) Metapath transformer encoder (VALU path): one wave per sequence.
//    lane = feature dim (D=32 == wave32). S=4 tokens, L=2 layers, H=4 heads.
// ---------------------------------------------------------------------------
__global__ __launch_bounds__(256) void k_metapath(
    const float* __restrict__ X, int in_dim,
    const float* __restrict__ Wemb, const float* __restrict__ bemb,
    const float* __restrict__ ln1s, const float* __restrict__ ln1b,
    const float* __restrict__ Wq, const float* __restrict__ bq,
    const float* __restrict__ Wk, const float* __restrict__ bk,
    const float* __restrict__ Wv, const float* __restrict__ bv,
    const float* __restrict__ Wo, const float* __restrict__ bo,
    const float* __restrict__ ln2s, const float* __restrict__ ln2b,
    const float* __restrict__ Wf1, const float* __restrict__ bf1,
    const float* __restrict__ Wf2, const float* __restrict__ bf2,
    const float* __restrict__ flns, const float* __restrict__ flnb,
    const float* __restrict__ Wa, const float* __restrict__ ba,
    float* __restrict__ out) {
  const int lane = threadIdx.x & 31;
  const int seq = blockIdx.x * 8 + (threadIdx.x >> 5);
  const float* xr = X + (size_t)seq * 4 * in_dim;

  float t[4];
#pragma unroll
  for (int s = 0; s < 4; ++s) {
    float a = bemb[lane];
    for (int k = 0; k < in_dim; ++k)
      a = fmaf(xr[s * in_dim + k], Wemb[k * 32 + lane], a);
    t[s] = a;
  }

  for (int l = 0; l < 2; ++l) {
    const float* wq = Wq + l * 1024;
    const float* wk = Wk + l * 1024;
    const float* wv = Wv + l * 1024;
    const float* wo = Wo + l * 1024;
    float y[4];
#pragma unroll
    for (int s = 0; s < 4; ++s) {  // LN1
      float m = wave_sum(t[s]) * 0.03125f;
      float d = t[s] - m;
      float v = wave_sum(d * d) * 0.03125f;
      y[s] = d * rsqrtf(v + 1e-5f) * ln1s[l * 32 + lane] + ln1b[l * 32 + lane];
    }
    float q[4], kk[4], vv[4];
#pragma unroll
    for (int s = 0; s < 4; ++s) {
      q[s] = bq[l * 32 + lane]; kk[s] = bk[l * 32 + lane]; vv[s] = bv[l * 32 + lane];
    }
    for (int k = 0; k < 32; ++k) {
      float a = wq[k * 32 + lane], b = wk[k * 32 + lane], c = wv[k * 32 + lane];
#pragma unroll
      for (int s = 0; s < 4; ++s) {
        float yk = __shfl(y[s], k, 32);
        q[s] = fmaf(yk, a, q[s]); kk[s] = fmaf(yk, b, kk[s]); vv[s] = fmaf(yk, c, vv[s]);
      }
    }
    float ctx[4];
#pragma unroll
    for (int s = 0; s < 4; ++s) {  // attention per head (DK=8 lanes)
      float sc[4];
#pragma unroll
      for (int s2 = 0; s2 < 4; ++s2)
        sc[s2] = head_sum(q[s] * kk[s2]) * 0.35355339059f;  // 1/sqrt(8)
      float mx = fmaxf(fmaxf(sc[0], sc[1]), fmaxf(sc[2], sc[3]));
      float e0 = expf(sc[0] - mx), e1 = expf(sc[1] - mx);
      float e2 = expf(sc[2] - mx), e3 = expf(sc[3] - mx);
      float inv = 1.0f / (e0 + e1 + e2 + e3);
      ctx[s] = (e0 * vv[0] + e1 * vv[1] + e2 * vv[2] + e3 * vv[3]) * inv;
    }
    float o[4];
#pragma unroll
    for (int s = 0; s < 4; ++s) o[s] = bo[l * 32 + lane];
    for (int k = 0; k < 32; ++k) {
      float w = wo[k * 32 + lane];
#pragma unroll
      for (int s = 0; s < 4; ++s) o[s] = fmaf(__shfl(ctx[s], k, 32), w, o[s]);
    }
#pragma unroll
    for (int s = 0; s < 4; ++s) t[s] += o[s];
#pragma unroll
    for (int s = 0; s < 4; ++s) {  // LN2
      float m = wave_sum(t[s]) * 0.03125f;
      float d = t[s] - m;
      float v = wave_sum(d * d) * 0.03125f;
      y[s] = d * rsqrtf(v + 1e-5f) * ln2s[l * 32 + lane] + ln2b[l * 32 + lane];
    }
    float h1[4], h2[4];
#pragma unroll
    for (int s = 0; s < 4; ++s) {
      h1[s] = bf1[l * 64 + lane]; h2[s] = bf1[l * 64 + 32 + lane];
    }
    const float* wf1 = Wf1 + l * 2048;
    const float* wf2 = Wf2 + l * 2048;
    for (int k = 0; k < 32; ++k) {
      float a = wf1[k * 64 + lane], b = wf1[k * 64 + 32 + lane];
#pragma unroll
      for (int s = 0; s < 4; ++s) {
        float yk = __shfl(y[s], k, 32);
        h1[s] = fmaf(yk, a, h1[s]); h2[s] = fmaf(yk, b, h2[s]);
      }
    }
#pragma unroll
    for (int s = 0; s < 4; ++s) { h1[s] = gelu_tanh(h1[s]); h2[s] = gelu_tanh(h2[s]); }
    float f[4];
#pragma unroll
    for (int s = 0; s < 4; ++s) f[s] = bf2[l * 32 + lane];
    for (int k = 0; k < 32; ++k) {
      float a = wf2[k * 32 + lane], b = wf2[(k + 32) * 32 + lane];
#pragma unroll
      for (int s = 0; s < 4; ++s) {
        f[s] = fmaf(__shfl(h1[s], k, 32), a, f[s]);
        f[s] = fmaf(__shfl(h2[s], k, 32), b, f[s]);
      }
    }
#pragma unroll
    for (int s = 0; s < 4; ++s) t[s] += f[s];
  }
  float o[4];
#pragma unroll
  for (int s = 0; s < 4; ++s) {  // final LN
    float m = wave_sum(t[s]) * 0.03125f;
    float d = t[s] - m;
    float v = wave_sum(d * d) * 0.03125f;
    o[s] = d * rsqrtf(v + 1e-5f) * flns[lane] + flnb[lane];
  }
  // attention pooling over the 3 neighbors
  float wan = Wa[lane], wag = Wa[32 + lane], bb = ba[0];
  float lg1 = wave_sum(o[0] * wan + o[1] * wag) + bb;
  float lg2 = wave_sum(o[0] * wan + o[2] * wag) + bb;
  float lg3 = wave_sum(o[0] * wan + o[3] * wag) + bb;
  float mx = fmaxf(lg1, fmaxf(lg2, lg3));
  float e1 = expf(lg1 - mx), e2 = expf(lg2 - mx), e3 = expf(lg3 - mx);
  float inv = 1.0f / (e1 + e2 + e3);
  out[(size_t)seq * 32 + lane] = o[0] + (e1 * o[1] + e2 * o[2] + e3 * o[3]) * inv;
}

// ---------------------------------------------------------------------------
// 3) Build dm (N x 64) = concat of metapath outputs; write hop-0 slot of both
//    feature stacks (f32) plus transposed bf16 copy for the GEMM B-operand.
// ---------------------------------------------------------------------------
__global__ __launch_bounds__(256) void k_build_dm(
    const float* __restrict__ encDrDi, const float* __restrict__ encDrG,
    const float* __restrict__ encMDi, const float* __restrict__ encMG,
    float* __restrict__ featsUp, float* __restrict__ featsDn,
    __bf16* __restrict__ xT0) {
  int idx = blockIdx.x * blockDim.x + threadIdx.x;  // NN*64
  int row = idx >> 6, c = idx & 63;
  float v;
  if (row < NDRUG)
    v = (c < 32) ? encDrDi[row * 32 + c] : encDrG[row * 32 + (c - 32)];
  else {
    int r = row - NDRUG;
    v = (c < 32) ? encMDi[r * 32 + c] : encMG[r * 32 + (c - 32)];
  }
  featsUp[row * 256 + c] = v;  // hop 0 slot (stride 4*64 floats per node)
  featsDn[row * 256 + c] = v;
  xT0[(size_t)c * NN + row] = (__bf16)v;
}

// ---------------------------------------------------------------------------
// 4a) Hop GEMM, 4-way K-split, 2x-unrolled ping-pong pipeline (no register
//     rotation copies):  partial[split] (8192x64 f32) = A[:,kseg] * x[kseg,:].
//     One wave per (32-row strip, K-segment); 1024 waves total.  The 4 waves
//     of a block share one K-segment and 4 adjacent strips (B reuse in L0/L2).
// ---------------------------------------------------------------------------
__global__ __launch_bounds__(128) void k_hop_gemm(
    const __bf16* __restrict__ A, const __bf16* __restrict__ xT,
    float* __restrict__ partial /* [KSPLIT][NN][64] */) {
  const int lane = threadIdx.x & 31;
  const int wid = threadIdx.x >> 5;
  const int lmod = lane & 15;
  const int lhalf = lane >> 4;
  const int split = blockIdx.x >> 6;                  // 0..3
  const int strip = (blockIdx.x & 63) * 4 + wid;      // 0..255
  const int row0 = strip * 32;
  const int kbeg = split * KSEG;
  const int kend = kbeg + KSEG;

  v8f acc[2][4];
  const v8f vz = {0.f, 0.f, 0.f, 0.f, 0.f, 0.f, 0.f, 0.f};
#pragma unroll
  for (int r = 0; r < 2; ++r)
#pragma unroll
    for (int t = 0; t < 4; ++t) acc[r][t] = vz;

  auto loadAB = [&](int k0, V16 a[2], v16bf b[4]) {
#pragma unroll
    for (int t = 0; t < 4; ++t)  // B tile: lane = output column, contiguous K
      b[t] = *(const v16bf*)(xT + (size_t)(t * 16 + lmod) * NN + k0 + lhalf * 16);
#pragma unroll
    for (int r = 0; r < 2; ++r) {
      const __bf16* ap = A + (size_t)(row0 + r * 16 + lmod) * NN + k0 + lhalf * 16;
      a[r].h[0] = *(const v8bf*)ap;
      a[r].h[1] = *(const v8bf*)(ap + 8);
      // pull the A stream toward the caches ~16 iterations ahead
      __builtin_prefetch((const void*)(ap + 512), 0, 1);
    }
  };
  auto mmac = [&](V16 a[2], v16bf b[4]) {
#pragma unroll
    for (int r = 0; r < 2; ++r)
#pragma unroll
      for (int t = 0; t < 4; ++t) acc[r][t] = wmma_bf16(a[r].v, b[t], acc[r][t]);
  };

  V16 a0[2], a1[2];
  v16bf b0[4], b1[4];
  loadAB(kbeg, a0, b0);
  loadAB(kbeg + 32, a1, b1);
  // steady state: consume buf0/buf1 and refill them in place (no copies);
  // loads cover kbeg+64 .. kend-32, computes cover kbeg .. kend-96.
  for (int k0 = kbeg + 64; k0 < kend; k0 += 64) {
    mmac(a0, b0);
    loadAB(k0, a0, b0);
    mmac(a1, b1);
    loadAB(k0 + 32, a1, b1);
  }
  mmac(a0, b0);  // kend-64
  mmac(a1, b1);  // kend-32

  float* dst = partial + (size_t)split * NN * 64;
#pragma unroll
  for (int r = 0; r < 2; ++r)
#pragma unroll
    for (int t = 0; t < 4; ++t) {
      int col = t * 16 + lmod;
      int rb = row0 + r * 16 + lhalf * 8;
#pragma unroll
      for (int m = 0; m < 8; ++m)
        dst[(size_t)(rb + m) * 64 + col] = acc[r][t][m];
    }
}

// 4b) Sum the K-split partials (fixed order -> deterministic), emit the f32
//     feats slot and the transposed bf16 operand for the next hop.
__global__ __launch_bounds__(256) void k_hop_finish(
    const float* __restrict__ partial, float* __restrict__ feats,
    __bf16* __restrict__ xTout, int hop) {
  int idx = blockIdx.x * blockDim.x + threadIdx.x;  // NN*64
  int row = idx >> 6, c = idx & 63;
  float v = partial[idx];
#pragma unroll
  for (int s = 1; s < KSPLIT; ++s) v += partial[(size_t)s * NN * 64 + idx];
  feats[row * 256 + hop * 64 + c] = v;
  xTout[(size_t)c * NN + row] = (__bf16)v;
}

// ---------------------------------------------------------------------------
// 5) Semantic attention (deterministic): per-row scalar w = tanh(z@Ws+bs)@qs,
//    fixed-order tree reduction to 4 group means, 2-way softmax, combine.
// ---------------------------------------------------------------------------
__global__ __launch_bounds__(256) void k_sem_w(
    const float* __restrict__ up, const float* __restrict__ dn,
    const float* __restrict__ Ws, const float* __restrict__ bs,
    const float* __restrict__ qs, float* __restrict__ wu_arr,
    float* __restrict__ wd_arr) {
  const int lane = threadIdx.x & 31;
  const int row = blockIdx.x * 8 + (threadIdx.x >> 5);
  float zu = up[(size_t)row * 32 + lane];
  float zd = dn[(size_t)row * 32 + lane];
  float au = bs[lane], ad = bs[lane];
  for (int k = 0; k < 32; ++k) {
    float w = Ws[k * 32 + lane];
    au = fmaf(__shfl(zu, k, 32), w, au);
    ad = fmaf(__shfl(zd, k, 32), w, ad);
  }
  float wu = wave_sum(tanhf(au) * qs[lane]);
  float wd = wave_sum(tanhf(ad) * qs[lane]);
  if (lane == 0) {
    wu_arr[row] = wu;
    wd_arr[row] = wd;
  }
}

__global__ __launch_bounds__(256) void k_sem_reduce(
    const float* __restrict__ wu_arr, const float* __restrict__ wd_arr,
    float* __restrict__ sums) {
  // blockIdx.x: 0=drug/up 1=drug/dn 2=meta/up 3=meta/dn
  __shared__ float red[256];
  const int tid = threadIdx.x;
  const float* src = (blockIdx.x & 1) ? wd_arr : wu_arr;
  const int beg = (blockIdx.x < 2) ? 0 : NDRUG;
  const int end = (blockIdx.x < 2) ? NDRUG : NN;
  float s = 0.0f;
  for (int i = beg + tid; i < end; i += 256) s += src[i];
  red[tid] = s;
  __syncthreads();
  for (int o = 128; o > 0; o >>= 1) {
    if (tid < o) red[tid] += red[tid + o];
    __syncthreads();
  }
  if (tid == 0) sums[((blockIdx.x < 2) ? 0 : 2) + (blockIdx.x & 1)] = red[0];
}

__global__ __launch_bounds__(256) void k_sem_combine(
    const float* __restrict__ up, const float* __restrict__ dn,
    const float* __restrict__ sums, float* __restrict__ nodef) {
  int idx = blockIdx.x * blockDim.x + threadIdx.x;  // NN*32
  int row = idx >> 5;
  int g = (row < NDRUG) ? 0 : 2;
  float n = (row < NDRUG) ? (float)NDRUG : (float)NMETA;
  float mu = sums[g] / n, md = sums[g + 1] / n;
  float m = fmaxf(mu, md);
  float eu = expf(mu - m), ed = expf(md - m);
  float inv = 1.0f / (eu + ed);
  nodef[idx] = (eu * up[idx] + ed * dn[idx]) * inv;
}

// ---------------------------------------------------------------------------
// 6) Prepare transposed bf16 MLP weights (Wm3 column-padded 3 -> 16).
// ---------------------------------------------------------------------------
__global__ __launch_bounds__(256) void k_prep_w(
    const float* __restrict__ Wm1, const float* __restrict__ Wm2,
    const float* __restrict__ Wm3, __bf16* __restrict__ W1T,
    __bf16* __restrict__ W2T, __bf16* __restrict__ W3T) {
  int i = blockIdx.x * blockDim.x + threadIdx.x;  // 8192 threads
  if (i < 8192) {  // W1T[o(128)][k(64)] = Wm1[k][o]
    int o = i >> 6, k = i & 63;
    W1T[i] = (__bf16)Wm1[k * 128 + o];
  }
  if (i < 8192) {  // W2T[o(64)][k(128)] = Wm2[k][o]
    int o = i >> 7, k = i & 127;
    W2T[i] = (__bf16)Wm2[k * 64 + o];
  }
  if (i < 1024) {  // W3T[o(16)][k(64)], rows 3..15 zero
    int o = i >> 6, k = i & 63;
    W3T[i] = (o < 3) ? (__bf16)Wm3[k * 3 + o] : (__bf16)0.0f;
  }
}

// ---------------------------------------------------------------------------
// 7) Fused edge MLP with WMMA: per wave, 16 edges: gather -> (16x64) ->
//    relu(@Wm1 16x128) -> relu(@Wm2 16x64) -> @Wm3 (16x3).  LDS round-trips
//    between C-layout and A-layout; intra-wave LDS deps via s_wait_dscnt.
// ---------------------------------------------------------------------------
__global__ __launch_bounds__(128) void k_edge_mlp(
    const int* __restrict__ trE, const int* __restrict__ vaE,
    const float* __restrict__ nodef, const __bf16* __restrict__ W1T,
    const __bf16* __restrict__ W2T, const __bf16* __restrict__ W3T,
    const float* __restrict__ bm1, const float* __restrict__ bm2,
    const float* __restrict__ bm3, float* __restrict__ out) {
  __shared__ __attribute__((aligned(32))) __bf16 ldsA[4][16 * 64];
  __shared__ __attribute__((aligned(32))) __bf16 ldsB[4][16 * 128];
  const int lane = threadIdx.x & 31;
  const int wid = threadIdx.x >> 5;
  const int lmod = lane & 15;
  const int lhalf = lane >> 4;
  const int ebase = (blockIdx.x * 4 + wid) * 16;

  {  // gather: lanes 0-15 fetch drug half, lanes 16-31 the meta half
    int e = ebase + lmod;
    int nodeRow;
    if (lhalf == 0)
      nodeRow = (e < ETR) ? trE[e] : vaE[e - ETR];
    else
      nodeRow = NDRUG + ((e < ETR) ? trE[ETR + e] : vaE[EVA + (e - ETR)]);
    const float* src = nodef + (size_t)nodeRow * 32;
    __bf16* dst = &ldsA[wid][lmod * 64 + lhalf * 32];
#pragma unroll
    for (int j = 0; j < 32; ++j) dst[j] = (__bf16)src[j];
  }
  asm volatile("s_wait_dscnt 0" ::: "memory");

  // layer 1: (16x64) @ (64x128) + bm1, relu
  v8f acc1[8];
#pragma unroll
  for (int t = 0; t < 8; ++t) {
    float b = bm1[t * 16 + lmod];
#pragma unroll
    for (int m = 0; m < 8; ++m) acc1[t][m] = b;
  }
#pragma unroll
  for (int kc = 0; kc < 2; ++kc) {
    v16bf a = loadA_rowmajor(&ldsA[wid][0], 64, kc * 32, lmod, lhalf);
#pragma unroll
    for (int t = 0; t < 8; ++t) {
      v16bf b = *(const v16bf*)(W1T + (t * 16 + lmod) * 64 + kc * 32 + lhalf * 16);
      acc1[t] = wmma_bf16(a, b, acc1[t]);
    }
  }
#pragma unroll
  for (int t = 0; t < 8; ++t) {
    int col = t * 16 + lmod, rb = lhalf * 8;
#pragma unroll
    for (int m = 0; m < 8; ++m)
      ldsB[wid][(rb + m) * 128 + col] = (__bf16)fmaxf(acc1[t][m], 0.0f);
  }
  asm volatile("s_wait_dscnt 0" ::: "memory");

  // layer 2: (16x128) @ (128x64) + bm2, relu
  v8f acc2[4];
#pragma unroll
  for (int t = 0; t < 4; ++t) {
    float b = bm2[t * 16 + lmod];
#pragma unroll
    for (int m = 0; m < 8; ++m) acc2[t][m] = b;
  }
#pragma unroll
  for (int kc = 0; kc < 4; ++kc) {
    v16bf a = loadA_rowmajor(&ldsB[wid][0], 128, kc * 32, lmod, lhalf);
#pragma unroll
    for (int t = 0; t < 4; ++t) {
      v16bf b = *(const v16bf*)(W2T + (t * 16 + lmod) * 128 + kc * 32 + lhalf * 16);
      acc2[t] = wmma_bf16(a, b, acc2[t]);
    }
  }
#pragma unroll
  for (int t = 0; t < 4; ++t) {
    int col = t * 16 + lmod, rb = lhalf * 8;
#pragma unroll
    for (int m = 0; m < 8; ++m)
      ldsA[wid][(rb + m) * 64 + col] = (__bf16)fmaxf(acc2[t][m], 0.0f);
  }
  asm volatile("s_wait_dscnt 0" ::: "memory");

  // layer 3: (16x64) @ (64x16 padded), only cols 0..2 valid
  v8f acc3;
  {
    float b = (lmod < 3) ? bm3[lmod] : 0.0f;
#pragma unroll
    for (int m = 0; m < 8; ++m) acc3[m] = b;
  }
#pragma unroll
  for (int kc = 0; kc < 2; ++kc) {
    v16bf a = loadA_rowmajor(&ldsA[wid][0], 64, kc * 32, lmod, lhalf);
    v16bf b = *(const v16bf*)(W3T + lmod * 64 + kc * 32 + lhalf * 16);
    acc3 = wmma_bf16(a, b, acc3);
  }
  if (lmod < 3) {
    int rb = lhalf * 8;
#pragma unroll
    for (int m = 0; m < 8; ++m)
      out[(size_t)(ebase + rb + m) * 3 + lmod] = acc3[m];
  }
}

// ---------------------------------------------------------------------------
// Host driver: workspace carving + full pipeline on `stream`.
// Needs ~305 MB of d_ws (two bf16 adjacency copies + K-split partials).
// ---------------------------------------------------------------------------
extern "C" void kernel_launch(void* const* d_in, const int* in_sizes, int n_in,
                              void* d_out, int out_size, void* d_ws,
                              size_t ws_size, hipStream_t stream) {
  (void)in_sizes; (void)n_in; (void)out_size; (void)ws_size;
  const float* adjDn = (const float*)d_in[0];
  const float* adjUp = (const float*)d_in[1];
  const float* hopsIn[4] = {(const float*)d_in[2], (const float*)d_in[3],
                            (const float*)d_in[4], (const float*)d_in[5]};
  const float* W80 = (const float*)d_in[6];  const float* b80 = (const float*)d_in[7];
  const float* W64 = (const float*)d_in[8];  const float* b64 = (const float*)d_in[9];
  const float* ln1s = (const float*)d_in[10]; const float* ln1b = (const float*)d_in[11];
  const float* Wq = (const float*)d_in[12];  const float* bq = (const float*)d_in[13];
  const float* Wk = (const float*)d_in[14];  const float* bk = (const float*)d_in[15];
  const float* Wv = (const float*)d_in[16];  const float* bv = (const float*)d_in[17];
  const float* Wo = (const float*)d_in[18];  const float* bo = (const float*)d_in[19];
  const float* ln2s = (const float*)d_in[20]; const float* ln2b = (const float*)d_in[21];
  const float* Wf1 = (const float*)d_in[22]; const float* bf1 = (const float*)d_in[23];
  const float* Wf2 = (const float*)d_in[24]; const float* bf2 = (const float*)d_in[25];
  const float* flns = (const float*)d_in[26]; const float* flnb = (const float*)d_in[27];
  const float* Wa = (const float*)d_in[28];  const float* ba = (const float*)d_in[29];
  const float* Ws = (const float*)d_in[30];  const float* bs = (const float*)d_in[31];
  const float* qs = (const float*)d_in[32];
  const float* Wm1 = (const float*)d_in[33]; const float* bm1 = (const float*)d_in[34];
  const float* Wm2 = (const float*)d_in[35]; const float* bm2 = (const float*)d_in[36];
  const float* Wm3 = (const float*)d_in[37]; const float* bm3 = (const float*)d_in[38];
  const int* trE = (const int*)d_in[39];
  const int* vaE = (const int*)d_in[40];

  char* ws = (char*)d_ws;
  size_t off = 0;
  auto take = [&](size_t bytes) -> char* {
    char* p = ws + off;
    off = (off + bytes + 255) & ~(size_t)255;
    return p;
  };
  __bf16* adjDnB = (__bf16*)take((size_t)NN * NN * 2);
  __bf16* adjUpB = (__bf16*)take((size_t)NN * NN * 2);
  float* enc[4];
  for (int i = 0; i < 4; ++i) enc[i] = (float*)take((size_t)NN * 32 * 4);
  float* featsUp = (float*)take((size_t)NN * 256 * 4);
  float* featsDn = (float*)take((size_t)NN * 256 * 4);
  float* partial = (float*)take((size_t)KSPLIT * NN * 64 * 4);
  __bf16* xT0 = (__bf16*)take((size_t)64 * NN * 2);
  __bf16* xTA = (__bf16*)take((size_t)64 * NN * 2);
  __bf16* xTB = (__bf16*)take((size_t)64 * NN * 2);
  float* up = (float*)take((size_t)NN * 32 * 4);
  float* dn = (float*)take((size_t)NN * 32 * 4);
  float* nodef = (float*)take((size_t)NN * 32 * 4);
  float* wu_arr = (float*)take((size_t)NN * 4);
  float* wd_arr = (float*)take((size_t)NN * 4);
  float* sums = (float*)take(256);
  __bf16* W1T = (__bf16*)take(128 * 64 * 2);
  __bf16* W2T = (__bf16*)take(64 * 128 * 2);
  __bf16* W3T = (__bf16*)take(16 * 64 * 2);

  // adjacency f32 -> bf16 (once per call; hop re-reads then hit the 192MB L2)
  k_f32_to_bf16<<<65536, 256, 0, stream>>>(adjDn, adjDnB, (size_t)NN * NN);
  k_f32_to_bf16<<<65536, 256, 0, stream>>>(adjUp, adjUpB, (size_t)NN * NN);

  // 4 metapath encoders on 80-dim hop inputs
  for (int i = 0; i < 4; ++i)
    k_metapath<<<1024, 256, 0, stream>>>(hopsIn[i], 80, W80, b80, ln1s, ln1b,
                                         Wq, bq, Wk, bk, Wv, bv, Wo, bo, ln2s,
                                         ln2b, Wf1, bf1, Wf2, bf2, flns, flnb,
                                         Wa, ba, enc[i]);

  // dm + hop-0 feature slots + transposed bf16 B-operand
  k_build_dm<<<2048, 256, 0, stream>>>(enc[0], enc[1], enc[2], enc[3], featsUp,
                                       featsDn, xT0);

  // 3 hops per adjacency: K-split WMMA GEMM + deterministic combine
  k_hop_gemm<<<256, 128, 0, stream>>>(adjDnB, xT0, partial);
  k_hop_finish<<<2048, 256, 0, stream>>>(partial, featsUp, xTA, 1);
  k_hop_gemm<<<256, 128, 0, stream>>>(adjDnB, xTA, partial);
  k_hop_finish<<<2048, 256, 0, stream>>>(partial, featsUp, xTB, 2);
  k_hop_gemm<<<256, 128, 0, stream>>>(adjDnB, xTB, partial);
  k_hop_finish<<<2048, 256, 0, stream>>>(partial, featsUp, xTA, 3);
  k_hop_gemm<<<256, 128, 0, stream>>>(adjUpB, xT0, partial);
  k_hop_finish<<<2048, 256, 0, stream>>>(partial, featsDn, xTA, 1);
  k_hop_gemm<<<256, 128, 0, stream>>>(adjUpB, xTA, partial);
  k_hop_finish<<<2048, 256, 0, stream>>>(partial, featsDn, xTB, 2);
  k_hop_gemm<<<256, 128, 0, stream>>>(adjUpB, xTB, partial);
  k_hop_finish<<<2048, 256, 0, stream>>>(partial, featsDn, xTA, 3);

  // metapath encoders on stacked hop features (64-dim)
  k_metapath<<<1024, 256, 0, stream>>>(featsUp, 64, W64, b64, ln1s, ln1b, Wq,
                                       bq, Wk, bk, Wv, bv, Wo, bo, ln2s, ln2b,
                                       Wf1, bf1, Wf2, bf2, flns, flnb, Wa, ba,
                                       up);
  k_metapath<<<1024, 256, 0, stream>>>(featsDn, 64, W64, b64, ln1s, ln1b, Wq,
                                       bq, Wk, bk, Wv, bv, Wo, bo, ln2s, ln2b,
                                       Wf1, bf1, Wf2, bf2, flns, flnb, Wa, ba,
                                       dn);

  // semantic attention (deterministic reduction)
  k_sem_w<<<1024, 256, 0, stream>>>(up, dn, Ws, bs, qs, wu_arr, wd_arr);
  k_sem_reduce<<<4, 256, 0, stream>>>(wu_arr, wd_arr, sums);
  k_sem_combine<<<1024, 256, 0, stream>>>(up, dn, sums, nodef);

  // fused edge MLP (WMMA)
  k_prep_w<<<32, 256, 0, stream>>>(Wm1, Wm2, Wm3, W1T, W2T, W3T);
  k_edge_mlp<<<ETOT / 64, 128, 0, stream>>>(trE, vaE, nodef, W1T, W2T, W3T,
                                            bm1, bm2, bm3, (float*)d_out);
}
